// MultiHeadSelfAttention_81415400063753
// MI455X (gfx1250) — compile-verified, matrix operand order checked by reading, not run on hardware
//
#include <hip/hip_runtime.h>
#include <hip/hip_bf16.h>

typedef __attribute__((ext_vector_type(16))) __bf16 bf16x16;
typedef __attribute__((ext_vector_type(8)))  float  f32x8;

#define B_SZ   16
#define N_SEQ  4096
#define DIN    1024
#define NH     8
#define DKH    128
#define RANK_K 1024                       // sel_idx + 1 = 0.25*4096
#define NORM_F 0.08838834764831845f      // 1/sqrt(128)

// ---- helpers -------------------------------------------------------------

__device__ __forceinline__ __bf16 f2bf(float f) { return (__bf16)f; }

// K index of the first element of the VGPR-r pair for this half-wave
// (16-bit A/B layout, CDNA5 ISA §7.12.2)
__device__ __forceinline__ int kpair_base(int r, int half) {
  return (r < 4 ? 2 * r : 16 + 2 * (r - 4)) + half * 8;
}

// Fragment whose K-dimension is contiguous in memory (A rows / row-major B^T):
// lane holds K = k0+half*8 .. +7 and k0+16+half*8 .. +7  -> two float4 pairs.
__device__ __forceinline__ bf16x16 load_frag_k_contig(const float* rowptr,
                                                      int k0, int half) {
  const float4* p0 = (const float4*)(rowptr + k0 + half * 8);
  const float4* p1 = (const float4*)(rowptr + k0 + 16 + half * 8);
  float4 f0 = p0[0], f1 = p0[1], f2 = p1[0], f3 = p1[1];
  bf16x16 v;
  v[0]  = f2bf(f0.x); v[1]  = f2bf(f0.y); v[2]  = f2bf(f0.z); v[3]  = f2bf(f0.w);
  v[4]  = f2bf(f1.x); v[5]  = f2bf(f1.y); v[6]  = f2bf(f1.z); v[7]  = f2bf(f1.w);
  v[8]  = f2bf(f2.x); v[9]  = f2bf(f2.y); v[10] = f2bf(f2.z); v[11] = f2bf(f2.w);
  v[12] = f2bf(f3.x); v[13] = f2bf(f3.y); v[14] = f2bf(f3.z); v[15] = f2bf(f3.w);
  return v;
}

// Fragment whose K-dimension is strided (column of a row-major matrix):
// colbase already points at this lane's column; element k at colbase + k*ld.
__device__ __forceinline__ bf16x16 load_frag_k_strided(const float* colbase,
                                                       int ld, int k0, int half) {
  bf16x16 v;
#pragma unroll
  for (int r = 0; r < 8; ++r) {
    int kb = k0 + kpair_base(r, half);
    v[2 * r]     = f2bf(colbase[(size_t)kb * ld]);
    v[2 * r + 1] = f2bf(colbase[(size_t)(kb + 1) * ld]);
  }
  return v;
}

__device__ __forceinline__ bf16x16 zero_frag() {
  bf16x16 v;
#pragma unroll
  for (int i = 0; i < 16; ++i) v[i] = (__bf16)0.0f;
  return v;
}

// order-preserving float<->u32 key (larger float -> larger key)
__device__ __forceinline__ unsigned tokey(float f) {
  unsigned u = __float_as_uint(f);
  return (u & 0x80000000u) ? ~u : (u | 0x80000000u);
}
__device__ __forceinline__ float fromkey(unsigned k) {
  unsigned u = (k & 0x80000000u) ? (k & 0x7fffffffu) : ~k;
  return __uint_as_float(u);
}

// ---- K1: q0 = x[:,0,:] @ Wq   (M=16 batches, N=1024, K=1024) -------------

__global__ __launch_bounds__(128)
void qproj_kernel(const float* __restrict__ x, const float* __restrict__ Wq,
                  float* __restrict__ q0) {
  int wave = threadIdx.x >> 5, lane = threadIdx.x & 31;
  int tile = blockIdx.x * 4 + wave;        // 64 tiles over 1024 columns
  int c0 = tile * 16;
  int li = lane & 15, half = lane >> 4;
  const float* arow = x + (size_t)li * N_SEQ * DIN;  // x[li, 0, :]
  const float* bcol = Wq + c0 + li;                  // column c0+li
  f32x8 acc = {};
#pragma unroll 2
  for (int k0 = 0; k0 < DIN; k0 += 32) {
    bf16x16 a  = load_frag_k_contig(arow, k0, half);
    bf16x16 bm = load_frag_k_strided(bcol, DIN, k0, half);
    acc = __builtin_amdgcn_wmma_f32_16x16x32_bf16(false, a, false, bm,
                                                  (short)0, acc, false, false);
  }
#pragma unroll
  for (int r = 0; r < 8; ++r) {
    int m = r + 8 * half;                  // batch
    q0[(size_t)m * DIN + c0 + li] = acc[r];
  }
}

// ---- K2: u[b,h,i] = Wk[i, h*128:(h+1)*128] . q0[b, h*128:(h+1)*128] ------

__global__ __launch_bounds__(256)
void uproj_kernel(const float* __restrict__ Wk, const float* __restrict__ q0,
                  float* __restrict__ u) {
  int t = blockIdx.x * 256 + threadIdx.x;  // over NH*DIN = 8192
  int b = blockIdx.y;
  int h = t >> 10, i = t & 1023;
  const float4* wrow = (const float4*)(Wk + (size_t)i * DIN + h * DKH);
  const float4* qv   = (const float4*)(q0 + (size_t)b * DIN + h * DKH);
  float acc = 0.f;
#pragma unroll 4
  for (int d = 0; d < DKH / 4; ++d) {
    float4 w = wrow[d], q = qv[d];
    acc += w.x * q.x + w.y * q.y + w.z * q.z + w.w * q.w;
  }
  u[((size_t)b * NH + h) * DIN + i] = acc;
}

// ---- K3: temp[b,h,n] = (x[b,n,:] . u[b,h,:]) * NORM  (WMMA, x pass 1) ----

__global__ __launch_bounds__(128)
void temp_kernel(const float* __restrict__ x, const float* __restrict__ u,
                 float* __restrict__ temp) {
  int wave = threadIdx.x >> 5, lane = threadIdx.x & 31;
  int b = blockIdx.y;
  int tile = blockIdx.x * 4 + wave;        // 256 tiles over N
  int n0 = tile * 16;
  int li = lane & 15, half = lane >> 4;
  const float* arow = x + ((size_t)b * N_SEQ + n0 + li) * DIN;  // x row
  const float* brow = u + ((size_t)b * NH + (li & 7)) * DIN;    // u row (head)
  bool bvalid = li < NH;
  f32x8 acc = {};
#pragma unroll 2
  for (int k0 = 0; k0 < DIN; k0 += 32) {
    __builtin_prefetch(arow + k0 + 256, 0, 0);
    bf16x16 a  = load_frag_k_contig(arow, k0, half);
    bf16x16 bm = bvalid ? load_frag_k_contig(brow, k0, half) : zero_frag();
    acc = __builtin_amdgcn_wmma_f32_16x16x32_bf16(false, a, false, bm,
                                                  (short)0, acc, false, false);
  }
  if (bvalid) {
#pragma unroll
    for (int r = 0; r < 8; ++r) {
      int m = r + 8 * half;                // x-row within tile
      temp[((size_t)b * NH + li) * N_SEQ + n0 + m] = acc[r] * NORM_F;
    }
  }
}

// ---- K4: exact 1024-th largest (radix select) + masked softmax, in place --

__global__ __launch_bounds__(256)
void select_softmax_kernel(float* __restrict__ temp) {
  __shared__ float    srow[N_SEQ];
  __shared__ int      hist[256];
  __shared__ unsigned sh_prefix;
  __shared__ int      sh_rank;
  __shared__ float    sred[256];
  int tid = threadIdx.x;
  float* row = temp + (size_t)blockIdx.x * N_SEQ;   // one (b,h) row
  for (int i = tid; i < N_SEQ; i += 256) srow[i] = row[i];
  __syncthreads();

  unsigned prefix = 0;
  int rank = RANK_K;                       // 1-indexed, descending
  for (int pass = 0; pass < 4; ++pass) {
    int shift = 24 - 8 * pass;
    hist[tid] = 0;
    __syncthreads();
    for (int i = tid; i < N_SEQ; i += 256) {
      unsigned key = tokey(srow[i]);
      bool ok = (pass == 0) || ((key >> (shift + 8)) == prefix);
      if (ok) atomicAdd(&hist[(key >> shift) & 255], 1);
    }
    __syncthreads();
    if (tid == 0) {
      int cum = 0, bin = 0;
      for (int bnum = 255; bnum >= 0; --bnum) {
        int c = hist[bnum];
        if (cum + c >= rank) { bin = bnum; break; }
        cum += c;
      }
      sh_prefix = (prefix << 8) | (unsigned)bin;
      sh_rank = rank - cum;
    }
    __syncthreads();
    prefix = sh_prefix;
    rank = sh_rank;
  }
  float thresh = fromkey(prefix);

  float lmax = -3.402823466e38f;
  for (int i = tid; i < N_SEQ; i += 256) {
    float zv = srow[i] >= thresh ? srow[i] : 0.f;
    lmax = fmaxf(lmax, zv);
  }
  sred[tid] = lmax; __syncthreads();
  for (int s = 128; s > 0; s >>= 1) {
    if (tid < s) sred[tid] = fmaxf(sred[tid], sred[tid + s]);
    __syncthreads();
  }
  float zmax = sred[0]; __syncthreads();

  float lsum = 0.f;
  for (int i = tid; i < N_SEQ; i += 256) {
    float zv = srow[i] >= thresh ? srow[i] : 0.f;
    lsum += __expf(zv - zmax);
  }
  sred[tid] = lsum; __syncthreads();
  for (int s = 128; s > 0; s >>= 1) {
    if (tid < s) sred[tid] += sred[tid + s];
    __syncthreads();
  }
  float inv = 1.f / sred[0];
  for (int i = tid; i < N_SEQ; i += 256) {
    float zv = srow[i] >= thresh ? srow[i] : 0.f;
    row[i] = __expf(zv - zmax) * inv;
  }
}

// ---- K5: y[b,h,:] = rel[b,h,:] @ X_b   (WMMA, x pass 2) ------------------

__global__ __launch_bounds__(128)
void yacc_kernel(const float* __restrict__ rel, const float* __restrict__ x,
                 float* __restrict__ y) {
  int wave = threadIdx.x >> 5, lane = threadIdx.x & 31;
  int b = blockIdx.y;
  int tile = blockIdx.x * 4 + wave;        // 64 tiles over DIN columns
  int n0 = tile * 16;
  int li = lane & 15, half = lane >> 4;
  const float* arow = rel + ((size_t)b * NH + (li & 7)) * N_SEQ; // rel row (head)
  const float* bcol = x + (size_t)b * N_SEQ * DIN + n0 + li;     // x column
  bool avalid = li < NH;
  f32x8 acc = {};
#pragma unroll 2
  for (int k0 = 0; k0 < N_SEQ; k0 += 32) {
    __builtin_prefetch(bcol + (size_t)(k0 + 32) * DIN, 0, 0);
    bf16x16 a  = avalid ? load_frag_k_contig(arow, k0, half) : zero_frag();
    bf16x16 bm = load_frag_k_strided(bcol, DIN, k0, half);
    acc = __builtin_amdgcn_wmma_f32_16x16x32_bf16(false, a, false, bm,
                                                  (short)0, acc, false, false);
  }
  if (half == 0) {                          // rows m=0..7 are the real heads
#pragma unroll
    for (int r = 0; r < 8; ++r) {
      y[((size_t)b * NH + r) * DIN + n0 + li] = acc[r];
    }
  }
}

// ---- K6: out[b, c] = y[b, c>>7, :] @ Wv[:, c]   (M=16 batches, WMMA) -----

__global__ __launch_bounds__(128)
void out_kernel(const float* __restrict__ y, const float* __restrict__ Wv,
                float* __restrict__ out) {
  int wave = threadIdx.x >> 5, lane = threadIdx.x & 31;
  int tile = blockIdx.x * 4 + wave;        // 64 tiles over 1024 output cols
  int c0 = tile * 16;
  int h = c0 >> 7;                          // head owning this column tile
  int li = lane & 15, half = lane >> 4;
  const float* arow = y + ((size_t)li * NH + h) * DIN;  // A[m=batch li][k]
  const float* bcol = Wv + c0 + li;                     // column c0+li
  f32x8 acc = {};
#pragma unroll 2
  for (int k0 = 0; k0 < DIN; k0 += 32) {
    bf16x16 a  = load_frag_k_contig(arow, k0, half);
    bf16x16 bm = load_frag_k_strided(bcol, DIN, k0, half);
    acc = __builtin_amdgcn_wmma_f32_16x16x32_bf16(false, a, false, bm,
                                                  (short)0, acc, false, false);
  }
#pragma unroll
  for (int r = 0; r < 8; ++r) {
    int m = r + 8 * half;                   // batch
    out[(size_t)m * DIN + c0 + li] = acc[r];
  }
}

// ---- launcher ------------------------------------------------------------

extern "C" void kernel_launch(void* const* d_in, const int* in_sizes, int n_in,
                              void* d_out, int out_size, void* d_ws, size_t ws_size,
                              hipStream_t stream) {
  (void)in_sizes; (void)n_in; (void)out_size; (void)ws_size;
  const float* x  = (const float*)d_in[0];
  const float* Wq = (const float*)d_in[1];
  const float* Wk = (const float*)d_in[2];
  const float* Wv = (const float*)d_in[3];

  float* ws   = (float*)d_ws;
  float* q0   = ws;                         // 16*1024            = 16384
  float* u    = q0 + 16 * 1024;             // 16*8*1024          = 131072
  float* temp = u + 16 * 8 * 1024;          // 16*8*4096          = 524288
  float* y    = temp + 16 * 8 * 4096;       // 16*8*1024          = 131072
                                            // total 3.2 MB of d_ws

  hipLaunchKernelGGL(qproj_kernel, dim3(16), dim3(128), 0, stream, x, Wq, q0);
  hipLaunchKernelGGL(uproj_kernel, dim3(32, 16), dim3(256), 0, stream, Wk, q0, u);
  hipLaunchKernelGGL(temp_kernel, dim3(64, 16), dim3(128), 0, stream, x, u, temp);
  hipLaunchKernelGGL(select_softmax_kernel, dim3(B_SZ * NH), dim3(256), 0, stream, temp);
  hipLaunchKernelGGL(yacc_kernel, dim3(16, 16), dim3(128), 0, stream, temp, x, y);
  hipLaunchKernelGGL(out_kernel, dim3(16), dim3(128), 0, stream, y, Wv, (float*)d_out);
}
